// WanAttentionBlockWithCudaKernel_12171937317138
// MI455X (gfx1250) — compile-verified, matrix operand order checked by reading, not run on hardware
//
#include <hip/hip_runtime.h>
#include <hip/hip_bf16.h>
#include <stdint.h>

constexpr int S_    = 2560;
constexpr int DIMC  = 1536;
constexpr int HEADS_= 12;
constexpr int HD_   = 128;
constexpr int FFN_  = 8960;
constexpr int L2_   = 512;

typedef _Float16 v16h __attribute__((ext_vector_type(16)));
typedef _Float16 v8h  __attribute__((ext_vector_type(8)));
typedef float    v8f  __attribute__((ext_vector_type(8)));
typedef int      v8i  __attribute__((ext_vector_type(8)));

#define DEV __device__ __forceinline__

DEV float block_sum256(float v, float* sbuf) {
  int tid = threadIdx.x;
  sbuf[tid] = v; __syncthreads();
  for (int s = 128; s > 0; s >>= 1) { if (tid < s) sbuf[tid] += sbuf[tid + s]; __syncthreads(); }
  float r = sbuf[0]; __syncthreads(); return r;
}
DEV float block_max256(float v, float* sbuf) {
  int tid = threadIdx.x;
  sbuf[tid] = v; __syncthreads();
  for (int s = 128; s > 0; s >>= 1) { if (tid < s) sbuf[tid] = fmaxf(sbuf[tid], sbuf[tid + s]); __syncthreads(); }
  float r = sbuf[0]; __syncthreads(); return r;
}
DEV float block_min256(float v, float* sbuf) {
  int tid = threadIdx.x;
  sbuf[tid] = v; __syncthreads();
  for (int s = 128; s > 0; s >>= 1) { if (tid < s) sbuf[tid] = fminf(sbuf[tid], sbuf[tid + s]); __syncthreads(); }
  float r = sbuf[0]; __syncthreads(); return r;
}

__global__ void k_em_add(const float* __restrict__ mod, const float* __restrict__ e,
                         float* __restrict__ em) {
  int i = blockIdx.x * 256 + threadIdx.x;
  if (i < 6 * DIMC) em[i] = mod[i] + e[i];
}

__global__ __launch_bounds__(256) void k_quant_w(
    const float* __restrict__ w0, const float* __restrict__ w1, const float* __restrict__ w2,
    int8_t* __restrict__ qw, float* __restrict__ sw, float* __restrict__ zp) {
  __shared__ float sbuf[256];
  int row = blockIdx.x, which = blockIdx.y;
  const float* w  = (which == 0) ? w0 : (which == 1) ? w1 : w2;
  const float* wr = w + (size_t)row * DIMC;
  float mn = 3.0e38f, mx = -3.0e38f;
  for (int c = threadIdx.x; c < DIMC; c += 256) { float v = wr[c]; mn = fminf(mn, v); mx = fmaxf(mx, v); }
  mn = block_min256(mn, sbuf);
  mx = block_max256(mx, sbuf);
  float s = fmaxf((mx - mn) / 255.0f, 1e-8f);
  float z = rintf(-128.0f - mn / s);
  int8_t* q = qw + ((size_t)which * DIMC + row) * DIMC;
  for (int c = threadIdx.x; c < DIMC; c += 256) {
    float qv = fminf(fmaxf(rintf(wr[c] / s) + z, -128.0f), 127.0f);
    q[c] = (int8_t)qv;
  }
  if (threadIdx.x == 0) { sw[which * DIMC + row] = s; zp[which * DIMC + row] = z; }
}

__global__ __launch_bounds__(256) void k_ln_mod_quant(
    const float* __restrict__ x, const float* __restrict__ em,
    int8_t* __restrict__ qx, float* __restrict__ sx, float* __restrict__ sumq) {
  __shared__ float sbuf[256];
  int t = blockIdx.x, tid = threadIdx.x;
  const float* xr = x + (size_t)t * DIMC;
  float r[6], s1 = 0.f, s2 = 0.f;
#pragma unroll
  for (int i = 0; i < 6; ++i) { float v = xr[tid + i * 256]; r[i] = v; s1 += v; s2 += v * v; }
  float mean = block_sum256(s1, sbuf) * (1.0f / DIMC);
  float var  = block_sum256(s2, sbuf) * (1.0f / DIMC) - mean * mean;
  float inv  = rsqrtf(var + 1e-6f);
  float amax = 0.f;
#pragma unroll
  for (int i = 0; i < 6; ++i) {
    int c = tid + i * 256;
    float h = (r[i] - mean) * inv * (1.0f + em[DIMC + c]) + em[c];
    r[i] = h; amax = fmaxf(amax, fabsf(h));
  }
  amax = block_max256(amax, sbuf);
  float s = fmaxf(amax / 127.0f, 1e-8f);
  float qs = 0.f;
#pragma unroll
  for (int i = 0; i < 6; ++i) {
    float q = fminf(fmaxf(rintf(r[i] / s), -127.0f), 127.0f);
    qs += q; qx[(size_t)t * DIMC + tid + i * 256] = (int8_t)q;
  }
  qs = block_sum256(qs, sbuf);
  if (tid == 0) { sx[t] = s; sumq[t] = qs; }
}

__global__ __launch_bounds__(256) void k_gemm_i8(
    const int8_t* __restrict__ qx, const int8_t* __restrict__ qw_all,
    const float* __restrict__ sw_all, const float* __restrict__ zp_all,
    const float* __restrict__ sx, const float* __restrict__ sumq,
    const float* __restrict__ bq, const float* __restrict__ bk, const float* __restrict__ bv,
    float* __restrict__ out_all) {
  constexpr int BM = 64, BN = 128, BK = 128, LDA = 160, LDB = 160;
  __shared__ int8_t Al[BM * LDA];
  __shared__ int8_t Bl[BN * LDB];
  int which = blockIdx.z;
  const int8_t* qw = qw_all + (size_t)which * DIMC * DIMC;
  const float*  sw = sw_all + which * DIMC;
  const float*  zp = zp_all + which * DIMC;
  const float* bias = (which == 0) ? bq : (which == 1) ? bk : bv;
  float* out = out_all + (size_t)which * S_ * DIMC;
  int m0 = blockIdx.y * BM, n0 = blockIdx.x * BN;
  int wave = threadIdx.x >> 5, lane = threadIdx.x & 31;
  int wr = wave >> 2, wc = wave & 3;
  v8i acc[2][2];
#pragma unroll
  for (int a = 0; a < 2; ++a)
#pragma unroll
    for (int b = 0; b < 2; ++b) acc[a][b] = 0;

  for (int k0 = 0; k0 < DIMC; k0 += BK) {
    {
      int row = threadIdx.x >> 2, col = (threadIdx.x & 3) * 32;
      *(int4*)&Al[row * LDA + col]      = *(const int4*)&qx[(size_t)(m0 + row) * DIMC + k0 + col];
      *(int4*)&Al[row * LDA + col + 16] = *(const int4*)&qx[(size_t)(m0 + row) * DIMC + k0 + col + 16];
    }
    {
      int row = threadIdx.x >> 1, col = (threadIdx.x & 1) * 64;
#pragma unroll
      for (int j = 0; j < 4; ++j)
        *(int4*)&Bl[row * LDB + col + j * 16] =
            *(const int4*)&qw[(size_t)(n0 + row) * DIMC + k0 + col + j * 16];
    }
    if (k0 + BK < DIMC)
      __builtin_prefetch(&qw[(size_t)(n0 + (threadIdx.x >> 1)) * DIMC + k0 + BK], 0, 1);
    __syncthreads();
#pragma unroll
    for (int kc = 0; kc < 2; ++kc) {
      v8i af[2], bf[2];
#pragma unroll
      for (int mi = 0; mi < 2; ++mi)
        af[mi] = *(const v8i*)&Al[(wr * 32 + mi * 16 + (lane & 15)) * LDA + kc * 64 + (lane >> 4) * 32];
#pragma unroll
      for (int ni = 0; ni < 2; ++ni)
        bf[ni] = *(const v8i*)&Bl[(wc * 32 + ni * 16 + (lane & 15)) * LDB + kc * 64 + (lane >> 4) * 32];
#pragma unroll
      for (int mi = 0; mi < 2; ++mi)
#pragma unroll
        for (int ni = 0; ni < 2; ++ni)
          acc[mi][ni] = __builtin_amdgcn_wmma_i32_16x16x64_iu8(
              true, af[mi], true, bf[ni], acc[mi][ni], false, false);
    }
    __syncthreads();
  }
#pragma unroll
  for (int mi = 0; mi < 2; ++mi)
#pragma unroll
    for (int ni = 0; ni < 2; ++ni)
#pragma unroll
      for (int r = 0; r < 8; ++r) {
        int row = m0 + wr * 32 + mi * 16 + r + 8 * (lane >> 4);
        int col = n0 + wc * 32 + ni * 16 + (lane & 15);
        float v = ((float)acc[mi][ni][r] - zp[col] * sumq[row]) * (sx[row] * sw[col]) + bias[col];
        out[(size_t)row * DIMC + col] = v;
      }
}

__global__ __launch_bounds__(256) void k_rms_rope(
    const float* __restrict__ qkv, const float* __restrict__ gq, const float* __restrict__ gk,
    const float* __restrict__ freqs, const int* __restrict__ gs,
    _Float16* __restrict__ qh, _Float16* __restrict__ kh, _Float16* __restrict__ vh) {
  __shared__ float sbuf[256];
  int t = blockIdx.x, which = blockIdx.y, tid = threadIdx.x;
  const float* src = qkv + ((size_t)which * S_ + t) * DIMC;
  _Float16* dst = ((which == 0) ? qh : (which == 1) ? kh : vh) + (size_t)t * DIMC;
  if (which == 2) {
#pragma unroll
    for (int i = 0; i < 6; ++i) { int c = tid + i * 256; dst[c] = (_Float16)src[c]; }
    return;
  }
  const float* g = (which == 0) ? gq : gk;
  int c0 = tid * 6;
  float r[6], ss = 0.f;
#pragma unroll
  for (int i = 0; i < 6; ++i) { r[i] = src[c0 + i]; ss += r[i] * r[i]; }
  ss = block_sum256(ss, sbuf);
  float inv = rsqrtf(ss * (1.0f / DIMC) + 1e-6f);
  int f = gs[0], hh = gs[1], ww = gs[2];
  int fi = t / (hh * ww); int rem = t - fi * hh * ww; int hi = rem / ww; int wi = rem - hi * ww;
#pragma unroll
  for (int j = 0; j < 3; ++j) {
    int c = c0 + 2 * j;
    int d = c & (HD_ - 1);
    int i = d >> 1;
    int pos = (i < 22) ? fi : ((i < 43) ? hi : wi);
    const float* fc = &freqs[((size_t)pos * 64 + i) * 2];
    float cs = fc[0], sn = fc[1];
    float a = r[2 * j]     * inv * g[c];
    float b = r[2 * j + 1] * inv * g[c + 1];
    dst[c]     = (_Float16)(a * cs - b * sn);
    dst[c + 1] = (_Float16)(a * sn + b * cs);
  }
}

__global__ __launch_bounds__(256) void k_rms_f16(
    const float* __restrict__ src0, const float* __restrict__ gain,
    _Float16* __restrict__ dst0) {
  __shared__ float sbuf[256];
  int t = blockIdx.x, tid = threadIdx.x;
  const float* src = src0 + (size_t)t * DIMC;
  _Float16* dst = dst0 + (size_t)t * DIMC;
  if (gain == nullptr) {
#pragma unroll
    for (int i = 0; i < 6; ++i) { int c = tid + i * 256; dst[c] = (_Float16)src[c]; }
    return;
  }
  float r[6], ss = 0.f;
#pragma unroll
  for (int i = 0; i < 6; ++i) { int c = tid + i * 256; r[i] = src[c]; ss += r[i] * r[i]; }
  ss = block_sum256(ss, sbuf);
  float inv = rsqrtf(ss * (1.0f / DIMC) + 1e-6f);
#pragma unroll
  for (int i = 0; i < 6; ++i) { int c = tid + i * 256; dst[c] = (_Float16)(r[i] * inv * gain[c]); }
}

__global__ __launch_bounds__(256) void k_ln_ss(
    const float* __restrict__ x, const float* __restrict__ sc, const float* __restrict__ sh,
    int add_one, float* __restrict__ out) {
  __shared__ float sbuf[256];
  int t = blockIdx.x, tid = threadIdx.x;
  const float* xr = x + (size_t)t * DIMC;
  float r[6], s1 = 0.f, s2 = 0.f;
#pragma unroll
  for (int i = 0; i < 6; ++i) { float v = xr[tid + i * 256]; r[i] = v; s1 += v; s2 += v * v; }
  float mean = block_sum256(s1, sbuf) * (1.0f / DIMC);
  float var  = block_sum256(s2, sbuf) * (1.0f / DIMC) - mean * mean;
  float inv  = rsqrtf(var + 1e-6f);
#pragma unroll
  for (int i = 0; i < 6; ++i) {
    int c = tid + i * 256;
    out[(size_t)t * DIMC + c] = (r[i] - mean) * inv * ((add_one ? 1.0f : 0.0f) + sc[c]) + sh[c];
  }
}

template <bool AHALF, bool OHALF, bool DOGELU, bool RES>
__global__ __launch_bounds__(256) void k_gemm_f16(
    const void* __restrict__ Aptr, const float* __restrict__ W,
    const float* __restrict__ bias, const float* __restrict__ resid,
    const float* __restrict__ gain, void* __restrict__ Optr,
    int M, int N, int K) {
  constexpr int BM = 64, BN = 256, BK = 64, LDH = 80;
  __shared__ _Float16 Al[BM * LDH];
  __shared__ _Float16 Bl[BN * LDH];
  int m0 = blockIdx.y * BM, n0 = blockIdx.x * BN;
  int wave = threadIdx.x >> 5, lane = threadIdx.x & 31;
  int wr = wave >> 2, wc = wave & 3;
  v8f acc[2][4];
#pragma unroll
  for (int a = 0; a < 2; ++a)
#pragma unroll
    for (int b = 0; b < 4; ++b) acc[a][b] = 0.f;

  for (int k0 = 0; k0 < K; k0 += BK) {
#pragma unroll
    for (int s = 0; s < 2; ++s) {
      int idx = threadIdx.x + s * 256;
      int row = idx >> 3, col = (idx & 7) * 8;
      v8h h;
      if (AHALF) {
        h = *(const v8h*)&((const _Float16*)Aptr)[(size_t)(m0 + row) * K + k0 + col];
      } else {
        const float* A = (const float*)Aptr;
        float4 f0 = *(const float4*)&A[(size_t)(m0 + row) * K + k0 + col];
        float4 f1 = *(const float4*)&A[(size_t)(m0 + row) * K + k0 + col + 4];
        h[0] = (_Float16)f0.x; h[1] = (_Float16)f0.y; h[2] = (_Float16)f0.z; h[3] = (_Float16)f0.w;
        h[4] = (_Float16)f1.x; h[5] = (_Float16)f1.y; h[6] = (_Float16)f1.z; h[7] = (_Float16)f1.w;
      }
      *(v8h*)&Al[row * LDH + col] = h;
    }
#pragma unroll
    for (int s = 0; s < 8; ++s) {
      int idx = threadIdx.x + s * 256;
      int row = idx >> 3, col = (idx & 7) * 8;
      float4 f0 = *(const float4*)&W[(size_t)(n0 + row) * K + k0 + col];
      float4 f1 = *(const float4*)&W[(size_t)(n0 + row) * K + k0 + col + 4];
      v8h h;
      h[0] = (_Float16)f0.x; h[1] = (_Float16)f0.y; h[2] = (_Float16)f0.z; h[3] = (_Float16)f0.w;
      h[4] = (_Float16)f1.x; h[5] = (_Float16)f1.y; h[6] = (_Float16)f1.z; h[7] = (_Float16)f1.w;
      *(v8h*)&Bl[row * LDH + col] = h;
    }
    if (k0 + BK < K)
      __builtin_prefetch(&W[(size_t)(n0 + (threadIdx.x)) * K + k0 + BK], 0, 1);
    __syncthreads();
#pragma unroll
    for (int kc = 0; kc < 2; ++kc) {
      v16h af[2], bf[4];
#pragma unroll
      for (int mi = 0; mi < 2; ++mi)
        af[mi] = *(const v16h*)&Al[(wr * 32 + mi * 16 + (lane & 15)) * LDH + kc * 32 + (lane >> 4) * 16];
#pragma unroll
      for (int ni = 0; ni < 4; ++ni)
        bf[ni] = *(const v16h*)&Bl[(wc * 64 + ni * 16 + (lane & 15)) * LDH + kc * 32 + (lane >> 4) * 16];
#pragma unroll
      for (int mi = 0; mi < 2; ++mi)
#pragma unroll
        for (int ni = 0; ni < 4; ++ni)
          acc[mi][ni] = __builtin_amdgcn_wmma_f32_16x16x32_f16(
              false, af[mi], false, bf[ni], (short)0, acc[mi][ni], false, false);
    }
    __syncthreads();
  }

#pragma unroll
  for (int mi = 0; mi < 2; ++mi)
#pragma unroll
    for (int ni = 0; ni < 4; ++ni)
#pragma unroll
      for (int r = 0; r < 8; ++r) {
        int row = m0 + wr * 32 + mi * 16 + r + 8 * (lane >> 4);
        int col = n0 + wc * 64 + ni * 16 + (lane & 15);
        float v = acc[mi][ni][r] + bias[col];
        if (DOGELU) {
          float u = 0.7978845608028654f * (v + 0.044715f * v * v * v);
          v = 0.5f * v * (1.0f + tanhf(u));
        }
        if (RES) {
          float g = (gain != nullptr) ? gain[col] : 1.0f;
          v = resid[(size_t)row * N + col] + v * g;
        }
        if (OHALF) ((_Float16*)Optr)[(size_t)row * N + col] = (_Float16)v;
        else       ((float*)Optr)[(size_t)row * N + col] = v;
      }
}

__global__ __launch_bounds__(128) void k_attn(
    const _Float16* __restrict__ Q, const _Float16* __restrict__ Kb,
    const _Float16* __restrict__ Vb, int kvlen, float* __restrict__ Out) {
  __shared__ _Float16 Kl[64 * 128];
  __shared__ _Float16 Vt[128 * 64];
  __shared__ _Float16 Pl[4][16 * 64];
  const float SCALE = 0.08838834764831845f;
  int head = blockIdx.y;
  int q0 = blockIdx.x * 64;
  int wave = threadIdx.x >> 5, lane = threadIdx.x & 31;
  int qrow = q0 + wave * 16;
  const size_t hoff = (size_t)head * HD_;

  v16h qf[4];
#pragma unroll
  for (int kc = 0; kc < 4; ++kc)
    qf[kc] = *(const v16h*)&Q[(size_t)(qrow + (lane & 15)) * DIMC + hoff + kc * 32 + (lane >> 4) * 16];

  v8f o[8];
#pragma unroll
  for (int nt = 0; nt < 8; ++nt) o[nt] = 0.f;
  float mrow[8], lrow[8];
#pragma unroll
  for (int r = 0; r < 8; ++r) { mrow[r] = -1e30f; lrow[r] = 0.f; }

  for (int kv0 = 0; kv0 < kvlen; kv0 += 64) {
    {
      int r = threadIdx.x >> 1, cseg = (threadIdx.x & 1) * 64;
      const _Float16* src = &Kb[(size_t)(kv0 + r) * DIMC + hoff + cseg];
#pragma unroll
      for (int j = 0; j < 8; ++j)
        ((int4*)&Kl[r * 128 + cseg])[j] = ((const int4*)src)[j];
    }
    {
#pragma unroll
      for (int it = 0; it < 8; ++it) {
        int idx = threadIdx.x + it * 128;
        int r = idx >> 4;
        int cs = (idx & 15) * 8;
        v8h hv = *(const v8h*)&Vb[(size_t)(kv0 + r) * DIMC + hoff + cs];
#pragma unroll
        for (int j = 0; j < 8; ++j) Vt[(cs + j) * 64 + r] = hv[j];
      }
    }
    __syncthreads();

    v8f sc4[4];
#pragma unroll
    for (int st = 0; st < 4; ++st) sc4[st] = 0.f;
#pragma unroll
    for (int kc = 0; kc < 4; ++kc) {
#pragma unroll
      for (int st = 0; st < 4; ++st) {
        v16h b = *(const v16h*)&Kl[(st * 16 + (lane & 15)) * 128 + kc * 32 + (lane >> 4) * 16];
        sc4[st] = __builtin_amdgcn_wmma_f32_16x16x32_f16(
            false, qf[kc], false, b, (short)0, sc4[st], false, false);
      }
    }

    float p4[4][8];
#pragma unroll
    for (int r = 0; r < 8; ++r) {
      float v0 = sc4[0][r] * SCALE, v1 = sc4[1][r] * SCALE;
      float v2 = sc4[2][r] * SCALE, v3 = sc4[3][r] * SCALE;
      float mx = fmaxf(fmaxf(v0, v1), fmaxf(v2, v3));
#pragma unroll
      for (int msk = 1; msk < 16; msk <<= 1) mx = fmaxf(mx, __shfl_xor(mx, msk, 32));
      float mnew = fmaxf(mrow[r], mx);
      float alpha = __expf(mrow[r] - mnew);
      p4[0][r] = __expf(v0 - mnew); p4[1][r] = __expf(v1 - mnew);
      p4[2][r] = __expf(v2 - mnew); p4[3][r] = __expf(v3 - mnew);
      float rs = p4[0][r] + p4[1][r] + p4[2][r] + p4[3][r];
#pragma unroll
      for (int msk = 1; msk < 16; msk <<= 1) rs += __shfl_xor(rs, msk, 32);
      lrow[r] = lrow[r] * alpha + rs;
      mrow[r] = mnew;
#pragma unroll
      for (int nt = 0; nt < 8; ++nt) o[nt][r] *= alpha;
    }

    _Float16* P = &Pl[wave][0];
#pragma unroll
    for (int r = 0; r < 8; ++r) {
      int rr = r + 8 * (lane >> 4);
#pragma unroll
      for (int st = 0; st < 4; ++st)
        P[rr * 64 + st * 16 + (lane & 15)] = (_Float16)p4[st][r];
    }
    v16h pf[2];
#pragma unroll
    for (int kc2 = 0; kc2 < 2; ++kc2)
      pf[kc2] = *(const v16h*)&P[(lane & 15) * 64 + kc2 * 32 + (lane >> 4) * 16];
#pragma unroll
    for (int nt = 0; nt < 8; ++nt) {
#pragma unroll
      for (int kc2 = 0; kc2 < 2; ++kc2) {
        v16h vf = *(const v16h*)&Vt[(nt * 16 + (lane & 15)) * 64 + kc2 * 32 + (lane >> 4) * 16];
        o[nt] = __builtin_amdgcn_wmma_f32_16x16x32_f16(
            false, pf[kc2], false, vf, (short)0, o[nt], false, false);
      }
    }
    __syncthreads();
  }

#pragma unroll
  for (int r = 0; r < 8; ++r) {
    float invl = 1.0f / lrow[r];
    int row = qrow + r + 8 * (lane >> 4);
#pragma unroll
    for (int nt = 0; nt < 8; ++nt) {
      int col = (int)hoff + nt * 16 + (lane & 15);
      Out[(size_t)row * DIMC + col] = o[nt][r] * invl;
    }
  }
}

extern "C" void kernel_launch(void* const* d_in, const int* in_sizes, int n_in,
                              void* d_out, int out_size, void* d_ws, size_t ws_size,
                              hipStream_t stream) {
  (void)in_sizes; (void)n_in; (void)out_size; (void)ws_size;
  const float* x      = (const float*)d_in[0];
  const float* e      = (const float*)d_in[1];
  const int*   gs     = (const int*)d_in[3];
  const float* freqs  = (const float*)d_in[4];
  const float* ctx    = (const float*)d_in[5];
  const float* mod    = (const float*)d_in[7];
  const float* wq = (const float*)d_in[8],  *bq = (const float*)d_in[9];
  const float* wk = (const float*)d_in[10], *bk = (const float*)d_in[11];
  const float* wv = (const float*)d_in[12], *bv = (const float*)d_in[13];
  const float* wo = (const float*)d_in[14], *bo = (const float*)d_in[15];
  const float* gq = (const float*)d_in[16], *gk = (const float*)d_in[17];
  const float* n3w = (const float*)d_in[18], *n3b = (const float*)d_in[19];
  const float* cwq = (const float*)d_in[20], *cbq = (const float*)d_in[21];
  const float* cwk = (const float*)d_in[22], *cbk = (const float*)d_in[23];
  const float* cwv = (const float*)d_in[24], *cbv = (const float*)d_in[25];
  const float* cwo = (const float*)d_in[26], *cbo = (const float*)d_in[27];
  const float* cgq = (const float*)d_in[28], *cgk = (const float*)d_in[29];
  const float* w1 = (const float*)d_in[30], *b1 = (const float*)d_in[31];
  const float* w2 = (const float*)d_in[32], *b2 = (const float*)d_in[33];

  char* base = (char*)d_ws;
  size_t off = 0;
  auto alloc = [&](size_t bytes) -> char* {
    char* p = base + off;
    off = (off + bytes + 255) & ~(size_t)255;
    return p;
  };
  int8_t*    QX  = (int8_t*)   alloc((size_t)S_ * DIMC);
  float*     SX  = (float*)    alloc((size_t)S_ * 4);
  float*     SQ  = (float*)    alloc((size_t)S_ * 4);
  int8_t*    QW  = (int8_t*)   alloc((size_t)3 * DIMC * DIMC);
  float*     SW  = (float*)    alloc((size_t)3 * DIMC * 4);
  float*     ZP  = (float*)    alloc((size_t)3 * DIMC * 4);
  float*     EM  = (float*)    alloc((size_t)6 * DIMC * 4);
  float*     QKV = (float*)    alloc((size_t)3 * S_ * DIMC * 4);
  _Float16*  QH  = (_Float16*) alloc((size_t)S_ * DIMC * 2);
  _Float16*  KH  = (_Float16*) alloc((size_t)S_ * DIMC * 2);
  _Float16*  VH  = (_Float16*) alloc((size_t)S_ * DIMC * 2);
  float*     ATT = (float*)    alloc((size_t)S_ * DIMC * 4);
  float*     X1  = (float*)    alloc((size_t)S_ * DIMC * 4);
  float*     HX  = (float*)    alloc((size_t)S_ * DIMC * 4);
  float*     CKT = (float*)    alloc((size_t)L2_ * DIMC * 4);
  float*     CVT = (float*)    alloc((size_t)L2_ * DIMC * 4);
  _Float16*  CQH = (_Float16*) alloc((size_t)S_ * DIMC * 2);
  _Float16*  CKH = (_Float16*) alloc((size_t)L2_ * DIMC * 2);
  _Float16*  CVH = (_Float16*) alloc((size_t)L2_ * DIMC * 2);
  float*     X2  = (float*)    alloc((size_t)S_ * DIMC * 4);
  _Float16*  G1  = (_Float16*) alloc((size_t)S_ * FFN_ * 2);
  float*     OUT = (float*)d_out;

  k_em_add<<<(6 * DIMC + 255) / 256, 256, 0, stream>>>(mod, e, EM);
  k_quant_w<<<dim3(DIMC, 3), 256, 0, stream>>>(wq, wk, wv, QW, SW, ZP);
  k_ln_mod_quant<<<S_, 256, 0, stream>>>(x, EM, QX, SX, SQ);
  k_gemm_i8<<<dim3(DIMC / 128, S_ / 64, 3), 256, 0, stream>>>(
      QX, QW, SW, ZP, SX, SQ, bq, bk, bv, QKV);
  k_rms_rope<<<dim3(S_, 3), 256, 0, stream>>>(QKV, gq, gk, freqs, gs, QH, KH, VH);
  k_attn<<<dim3(S_ / 64, HEADS_), 128, 0, stream>>>(QH, KH, VH, S_, ATT);
  k_gemm_f16<false, false, false, true><<<dim3(DIMC / 256, S_ / 64), 256, 0, stream>>>(
      ATT, wo, bo, x, EM + 2 * DIMC, X1, S_, DIMC, DIMC);
  k_ln_ss<<<S_, 256, 0, stream>>>(X1, n3w, n3b, 0, HX);
  k_gemm_f16<false, false, false, false><<<dim3(DIMC / 256, S_ / 64), 256, 0, stream>>>(
      HX, cwq, cbq, nullptr, nullptr, QKV, S_, DIMC, DIMC);
  k_gemm_f16<false, false, false, false><<<dim3(DIMC / 256, L2_ / 64), 256, 0, stream>>>(
      ctx, cwk, cbk, nullptr, nullptr, CKT, L2_, DIMC, DIMC);
  k_gemm_f16<false, false, false, false><<<dim3(DIMC / 256, L2_ / 64), 256, 0, stream>>>(
      ctx, cwv, cbv, nullptr, nullptr, CVT, L2_, DIMC, DIMC);
  k_rms_f16<<<S_, 256, 0, stream>>>(QKV, cgq, CQH);
  k_rms_f16<<<L2_, 256, 0, stream>>>(CKT, cgk, CKH);
  k_rms_f16<<<L2_, 256, 0, stream>>>(CVT, nullptr, CVH);
  k_attn<<<dim3(S_ / 64, HEADS_), 128, 0, stream>>>(CQH, CKH, CVH, L2_, ATT);
  k_gemm_f16<false, false, false, true><<<dim3(DIMC / 256, S_ / 64), 256, 0, stream>>>(
      ATT, cwo, cbo, X1, nullptr, X2, S_, DIMC, DIMC);
  k_ln_ss<<<S_, 256, 0, stream>>>(X2, EM + 4 * DIMC, EM + 3 * DIMC, 1, HX);
  k_gemm_f16<false, true, true, false><<<dim3(FFN_ / 256, S_ / 64), 256, 0, stream>>>(
      HX, w1, b1, nullptr, nullptr, G1, S_, FFN_, DIMC);
  k_gemm_f16<true, false, false, true><<<dim3(DIMC / 256, S_ / 64), 256, 0, stream>>>(
      G1, w2, b2, X2, EM + 5 * DIMC, OUT, S_, DIMC, FFN_);
}
